// LSTM_2190433321774
// MI455X (gfx1250) — matrix operand order repaired and checked by reading, not compiled, not run on hardware
//
#include <hip/hip_runtime.h>

#define HID   512
#define IN    64
#define KTOT  576                 // HID + IN : augmented K per step
#define NKS   18                  // KTOT / 32 k-steps
#define NG    2048                // 4*HID gate rows
#define TT    256
#define NCLS  10
#define LDA   584                 // padded LDS row stride (292 dwords = 4*73 -> conflict-free)

typedef __attribute__((ext_vector_type(16))) __bf16 v16bf;
typedef __attribute__((ext_vector_type(8)))  float  v8f;

union Frag {
    uint4 u[2];
    v16bf v;
};

__device__ __forceinline__ float fast_sigmoid(float x) {
    return 1.0f / (1.0f + __expf(-x));
}
__device__ __forceinline__ float fast_tanh(float x) {
    float e = __expf(2.0f * x);
    return (e - 1.0f) / (e + 1.0f);
}

// ---------------------------------------------------------------------------
// Prep: build augmented bf16 weight Wcat[2048][576] = [W_hh | W_ih] and the
// fused fp32 bias (b_ih + b_hh) in workspace.
// ---------------------------------------------------------------------------
__global__ void lstm_prep_kernel(const float* __restrict__ Whh,
                                 const float* __restrict__ Wih,
                                 const float* __restrict__ bih,
                                 const float* __restrict__ bhh,
                                 __bf16* __restrict__ wcat,
                                 float*  __restrict__ bcomb) {
    int i = blockIdx.x * blockDim.x + threadIdx.x;
    if (i < NG * KTOT) {
        int row = i / KTOT, col = i % KTOT;
        float v = (col < HID) ? Whh[row * HID + col] : Wih[row * IN + (col - HID)];
        wcat[i] = (__bf16)v;
    }
    if (i < NG) bcomb[i] = bih[i] + bhh[i];
}

// ---------------------------------------------------------------------------
// Persistent LSTM kernel.
//   grid.x = B/16 batch tiles, block = 512 threads = 16 waves (wave32).
//   Wave w owns hidden slice [w*32, w*32+32): 4 gates x 2 col-halves
//   = 8 f32 16x16 accumulator tiles; K = 576 ([h|x]) as 18 WMMA k-steps.
//   A laundered 32-bit zero offset defeats LICM (no spills) while keeping
//   global_load codegen; sched_group_barriers pin a 1-k-step-lookahead
//   pipeline: [16 VMEM (B next)][2 DS (A next)][8 WMMA (current)].
// ---------------------------------------------------------------------------
__launch_bounds__(512, 1)
__global__ void lstm_persistent_kernel(const float*  __restrict__ x,
                                       const float*  __restrict__ h0,
                                       const float*  __restrict__ c0,
                                       const __bf16* __restrict__ Wcat,  // [2048][576] bf16
                                       const float*  __restrict__ bcomb, // [2048]
                                       const float*  __restrict__ Wlin,  // [10][512]
                                       const float*  __restrict__ blin,  // [10]
                                       float* __restrict__ out) {
    __shared__ __align__(16) __bf16 hx[16 * LDA];   // [h(512) | x(64)] padded rows, ~18.7 KB

    const int tid  = threadIdx.x;
    const int lane = tid & 31;
    const int wave = tid >> 5;            // 0..15
    const int bB   = blockIdx.x * 16;     // batch-row base

    const int nl = lane & 15;             // local N (and local M for A-frags)
    const int hi = lane >> 4;             // lane-half selector
    const int js = wave * 32;             // this wave's hidden slice base

    // --- stage h0 (fp32 -> bf16) into LDS ---
    for (int i = tid; i < 16 * HID; i += 512) {
        int r = i >> 9, c = i & (HID - 1);
        hx[r * LDA + c] = (__bf16)h0[(bB + r) * HID + c];
    }

    // --- per-lane fp32 cell state + fused biases ---
    // element (m, n): m = r + 8*hi (VGPR row), n = js + hf*16 + nl
    float cst[2][8];
    float bsv[4][2];
#pragma unroll
    for (int hf = 0; hf < 2; ++hf) {
        const int n = js + hf * 16 + nl;
#pragma unroll
        for (int r = 0; r < 8; ++r)
            cst[hf][r] = c0[(bB + r + 8 * hi) * HID + n];
#pragma unroll
        for (int g = 0; g < 4; ++g)
            bsv[g][hf] = bcomb[g * HID + n];
    }

    // per-tile weight-row byte offsets (tile i = g*2+hf); lane-dependent
    unsigned woff[8];
#pragma unroll
    for (int g = 0; g < 4; ++g)
#pragma unroll
        for (int hf = 0; hf < 2; ++hf)
            woff[g * 2 + hf] =
                (unsigned)(((g * HID + js + hf * 16 + nl) * KTOT + hi * 16) *
                           sizeof(__bf16));

    __syncthreads();

    for (int t = 0; t < TT; ++t) {
        // ---- stage x_t tile (16 x 64) fp32 -> bf16 into hx[:, 512..575] ----
#pragma unroll
        for (int i = tid; i < 16 * IN; i += 512) {
            int r = i >> 6, c = i & (IN - 1);
            hx[r * LDA + HID + c] = (__bf16)x[(bB + r) * (TT * IN) + t * IN + c];
        }
        // prefetch next step's x tile into near cache (global_prefetch_b8)
        if (t + 1 < TT && tid < 16)
            __builtin_prefetch(&x[(bB + tid) * (TT * IN) + (t + 1) * IN], 0, 3);

        __syncthreads();   // x staged + previous h visible, prior reads done

        // Laundered zero: weight addresses look t-variant (no LICM hoist ->
        // no spills) but stay Wcat-derived (addrspace inference -> global_load).
        unsigned zoff = 0;
        asm volatile("" : "+v"(zoff));

        v8f acc[8];
#pragma unroll
        for (int i = 0; i < 8; ++i)
            acc[i] = (v8f){0.f, 0.f, 0.f, 0.f, 0.f, 0.f, 0.f, 0.f};

        // B 32x16 bf16 frag: lane n=nl, K = ks*32 + hi*16 .. +15
        auto loadB = [&](Frag* d, int ks) {
#pragma unroll
            for (int i = 0; i < 8; ++i) {
                const __bf16* p =
                    (const __bf16*)((const char*)Wcat + (woff[i] + zoff)) + ks * 32;
                d[i].u[0] = *(const uint4*)(p);
                d[i].u[1] = *(const uint4*)(p + 8);
            }
        };
        // A 16x32 bf16 frag: lane m=nl, K = k0 + hi*16 .. +15
        auto loadA = [&](Frag& a, int ks) {
            const int k0 = ks * 32;
            a.u[0] = *(const uint4*)&hx[nl * LDA + k0 + hi * 16];
            a.u[1] = *(const uint4*)&hx[nl * LDA + k0 + hi * 16 + 8];
        };
        auto wmma8 = [&](const Frag& a, const Frag* b) {
#pragma unroll
            for (int i = 0; i < 8; ++i)
                acc[i] = __builtin_amdgcn_wmma_f32_16x16x32_bf16(
                    false, a.v, false, b[i].v, (short)0, acc[i], false, false);
        };

        // ---- software-pipelined K loop: 1 k-step lookahead on A and B ----
        Frag b0[8], b1[8], a0, a1;
        loadB(b0, 0);
        loadA(a0, 0);
        __builtin_amdgcn_sched_group_barrier(0x020, 16, 0);  // prologue B loads
        __builtin_amdgcn_sched_group_barrier(0x100, 2,  0);  // prologue A loads
#pragma unroll
        for (int ks = 0; ks < NKS; ks += 2) {
            loadB(b1, ks + 1);
            loadA(a1, ks + 1);
            wmma8(a0, b0);
            __builtin_amdgcn_sched_group_barrier(0x020, 16, 0);  // B(ks+1)
            __builtin_amdgcn_sched_group_barrier(0x100, 2,  0);  // A(ks+1)
            __builtin_amdgcn_sched_group_barrier(0x008, 8,  0);  // WMMA(ks)
            if (ks + 2 < NKS) {
                loadB(b0, ks + 2);
                loadA(a0, ks + 2);
                __builtin_amdgcn_sched_group_barrier(0x020, 16, 0);  // B(ks+2)
                __builtin_amdgcn_sched_group_barrier(0x100, 2,  0);  // A(ks+2)
            }
            wmma8(a1, b1);
            __builtin_amdgcn_sched_group_barrier(0x008, 8, 0);       // WMMA(ks+1)
        }

        __syncthreads();   // all WMMA reads of hx complete

        // ---- fused LSTM cell update: identical (lane,vgpr) mapping in all
        //      four gate tiles -> purely lane-local elementwise math ----
#pragma unroll
        for (int hf = 0; hf < 2; ++hf) {
            const int n = js + hf * 16 + nl;
#pragma unroll
            for (int r = 0; r < 8; ++r) {
                float iv = fast_sigmoid(acc[0 * 2 + hf][r] + bsv[0][hf]);
                float fv = fast_sigmoid(acc[1 * 2 + hf][r] + bsv[1][hf]);
                float gv = fast_tanh   (acc[2 * 2 + hf][r] + bsv[2][hf]);
                float ov = fast_sigmoid(acc[3 * 2 + hf][r] + bsv[3][hf]);
                float cn = fv * cst[hf][r] + iv * gv;
                cst[hf][r] = cn;
                float hn = ov * fast_tanh(cn);
                hx[(r + 8 * hi) * LDA + n] = (__bf16)hn;   // m = r+8*hi
            }
        }
        // next iteration's __syncthreads() publishes h before any reads
    }

    __syncthreads();

    // ---- final classifier: preds[16,10] = h_T @ Wlin^T + blin (tiny) ----
    if (tid < 16 * NCLS) {
        const int bl  = tid / NCLS;
        const int cls = tid % NCLS;
        float s = blin[cls];
        for (int k = 0; k < HID; ++k)
            s += (float)hx[bl * LDA + k] * Wlin[cls * HID + k];
        out[(bB + bl) * NCLS + cls] = s;
    }
}

// ---------------------------------------------------------------------------
extern "C" void kernel_launch(void* const* d_in, const int* in_sizes, int n_in,
                              void* d_out, int out_size, void* d_ws, size_t ws_size,
                              hipStream_t stream) {
    const float* x    = (const float*)d_in[0];
    const float* h0   = (const float*)d_in[1];
    const float* c0   = (const float*)d_in[2];
    const float* Wih  = (const float*)d_in[3];
    const float* Whh  = (const float*)d_in[4];
    const float* bih  = (const float*)d_in[5];
    const float* bhh  = (const float*)d_in[6];
    const float* Wlin = (const float*)d_in[7];
    const float* blin = (const float*)d_in[8];
    float* out = (float*)d_out;

    // workspace layout: bf16 Wcat [2048][576] | fp32 fused bias [2048]
    char* ws = (char*)d_ws;
    __bf16* wcat  = (__bf16*)(ws);
    float*  bcomb = (float*)(ws + (size_t)NG * KTOT * 2);

    const int prepN = NG * KTOT;
    lstm_prep_kernel<<<(prepN + 255) / 256, 256, 0, stream>>>(
        Whh, Wih, bih, bhh, wcat, bcomb);

    lstm_persistent_kernel<<<1024 / 16, 512, 0, stream>>>(
        x, h0, c0, wcat, bcomb, Wlin, blin, out);
}
